// DotProductAttention_38860864094295
// MI455X (gfx1250) — matrix operand order repaired and checked by reading, not compiled, not run on hardware
//
#include <hip/hip_runtime.h>

typedef __attribute__((ext_vector_type(16))) __bf16 v16bf;
typedef __attribute__((ext_vector_type(8)))  __bf16 bf16x8;
typedef __attribute__((ext_vector_type(4)))  __bf16 bf16x4;
typedef __attribute__((ext_vector_type(8)))  float  v8f;
typedef __attribute__((ext_vector_type(4)))  float  f4;

#define S_LEN 2048
#define NB 4
#define NH 16
#define DH 128
#define QTILE 128
#define KTILE 64
#define KPAD 136   // bf16 elems per ldsK row (128 + 8 pad) -> 272B, 16B aligned
#define VPAD 72    // bf16 elems per ldsVT / ldsP row (64 + 8 pad) -> 144B, 16B aligned
#define ROWSTRIDE (NB * NH * DH)   // floats between consecutive seq positions

// raw v_exp_f32 (exp2). Inputs here are always <= 0; hardware flush-to-zero for
// very negative inputs is exactly the behavior softmax wants (no libm guard code).
__device__ __forceinline__ float fast_exp2(float x) { return __builtin_amdgcn_exp2f(x); }

__device__ __forceinline__
void stage_fill(const float* __restrict__ Kg, const float* __restrict__ Vg,
                size_t bh_off, int kb, __bf16* bK, __bf16* bVT, int tid)
{
    // K tile [64 x 128] -> bf16 row-major (coalesced 16B reads)
    {
        const int dchunk = (tid & 31) * 4;
#pragma unroll
        for (int pass = 0; pass < 8; ++pass) {
            const int key = (tid >> 5) + 8 * pass;
            f4 x = *(const f4*)(Kg + (size_t)(kb + key) * ROWSTRIDE + bh_off + dchunk);
            bf16x4 w;
#pragma unroll
            for (int k = 0; k < 4; ++k) w[k] = (__bf16)x[k];
            *(bf16x4*)&bK[key * KPAD + dchunk] = w;
        }
    }
    // V tile transposed [128 x 64] via 4x4 register micro-transpose
    {
#pragma unroll
        for (int i = 0; i < 2; ++i) {
            const int task = tid + 256 * i;
            const int dt = (task & 31) * 4;
            const int kt = (task >> 5) * 4;
            f4 v4[4];
#pragma unroll
            for (int j = 0; j < 4; ++j)
                v4[j] = *(const f4*)(Vg + (size_t)(kb + kt + j) * ROWSTRIDE + bh_off + dt);
#pragma unroll
            for (int jd = 0; jd < 4; ++jd) {
                bf16x4 w;
#pragma unroll
                for (int jk = 0; jk < 4; ++jk) w[jk] = (__bf16)v4[jk][jd];
                *(bf16x4*)&bVT[(dt + jd) * VPAD + kt] = w;
            }
        }
    }
}

__global__ __launch_bounds__(256)
void fa_fwd_causal(const float* __restrict__ Qg, const float* __restrict__ Kg,
                   const float* __restrict__ Vg, float* __restrict__ Og)
{
    __shared__ __align__(16) __bf16 ldsK [2][KTILE * KPAD];   // ping-pong K tiles
    __shared__ __align__(16) __bf16 ldsVT[2][DH    * VPAD];   // ping-pong V^T tiles
    __shared__ __align__(16) __bf16 ldsP [8 * 16 * VPAD];     // per-wave P^T staging

    const int tid  = threadIdx.x;
    const int lane = tid & 31;
    const int wave = tid >> 5;
    const int ln   = lane & 15;   // fragment column
    const int hf   = lane >> 4;   // wave half

    const int qblk = blockIdx.x;
    const int bi   = blockIdx.y;
    const int hi   = blockIdx.z;

    const size_t bh_off = (size_t)(bi * NH + hi) * DH;
    const int qw   = qblk * QTILE + wave * 16;
    const int qrow = qw + ln;

    // Q^T fragments (WMMA B-matrix layout): lane holds q=ln, element e -> d = 32c + (e<8 ? e : e+8) + 8*hf
    v16bf qf[4];
    {
        const float* qp = Qg + (size_t)qrow * ROWSTRIDE + bh_off;
#pragma unroll
        for (int c = 0; c < 4; ++c) {
            const f4* p = (const f4*)(qp + 32 * c + 8 * hf);
#pragma unroll
            for (int j = 0; j < 4; ++j) {
                f4 x = p[j + (j & 2)];
#pragma unroll
                for (int k = 0; k < 4; ++k) qf[c][4 * j + k] = (__bf16)x[k];
            }
        }
    }

    const v8f vzero = {0.f, 0.f, 0.f, 0.f, 0.f, 0.f, 0.f, 0.f};
    v8f oacc[8];   // O^T accumulators: chunk c holds dims 16c + r + 8*hf, q = ln
#pragma unroll
    for (int c = 0; c < 8; ++c) oacc[c] = vzero;
    float m_run = -3.0e38f, l_run = 0.0f;
    // 1/sqrt(128) * log2(e): scores scaled directly into exp2 domain
    const float scale2 = 0.08838834764831845f * 1.4426950408889634f;

    __bf16* pbuf = ldsP + wave * (16 * VPAD);
    const int kmax = qblk * QTILE + QTILE;

    // prologue: fill stage 0 into buffer 0
    stage_fill(Kg, Vg, bh_off, 0, ldsK[0], ldsVT[0], tid);

    int p = 0;
    for (int kb = 0; kb < kmax; kb += KTILE, p ^= 1) {
        __syncthreads();   // fill(buf p) complete; all waves done computing stage kb-1

        // prefetch two stages ahead into cache (global_prefetch_b8)
        {
            const int kpf = kb + 2 * KTILE;
            if (kpf < S_LEN) {
                const size_t roff = (size_t)(kpf + (tid >> 2)) * ROWSTRIDE + bh_off;
                __builtin_prefetch((const char*)(Kg + roff) + (tid & 3) * 128, 0, 1);
                __builtin_prefetch((const char*)(Vg + roff) + (tid & 3) * 128, 0, 1);
            }
        }
        // fill next stage into the other buffer (overlaps with compute below)
        if (kb + KTILE < kmax)
            stage_fill(Kg, Vg, bh_off, kb + KTILE, ldsK[p ^ 1], ldsVT[p ^ 1], tid);

        if (kb > qw) continue;              // wave entirely above this key tile
        const bool diag = (qw < kb + KTILE);
        const __bf16* cK  = ldsK[p];
        const __bf16* cVT = ldsVT[p];

        // ---- S^T = K * Q^T  (4 subtiles of 16 keys, accumulate over 4 d-chunks of 32)
        v8f st[4];
#pragma unroll
        for (int t = 0; t < 4; ++t) {
            v8f acc = vzero;
#pragma unroll
            for (int c = 0; c < 4; ++c) {
                // A-frag (K): lane holds key row ln; element e -> d = 32c + (e<8 ? e : e+8) + 8*hf
                const __bf16* kr = &cK[(16 * t + ln) * KPAD + 32 * c + 8 * hf];
                bf16x8 lo = *(const bf16x8*)kr;
                bf16x8 hi = *(const bf16x8*)(kr + 16);
                v16bf a = __builtin_shufflevector(lo, hi, 0,1,2,3,4,5,6,7,8,9,10,11,12,13,14,15);
                acc = __builtin_amdgcn_wmma_f32_16x16x32_bf16(
                          false, a, false, qf[c], (short)0, acc, false, false);
            }
            st[t] = acc * scale2;
            if (diag) {
#pragma unroll
                for (int r = 0; r < 8; ++r)
                    if (kb + 16 * t + 8 * hf + r > qrow) st[t][r] = -1.0e30f;
            }
        }

        // ---- online softmax (exp2 domain) per q column = per lane
        float mloc = -3.0e38f;
#pragma unroll
        for (int t = 0; t < 4; ++t)
#pragma unroll
            for (int r = 0; r < 8; ++r) mloc = fmaxf(mloc, st[t][r]);
        mloc = fmaxf(mloc, __shfl_xor(mloc, 16, 32));

        // wave-uniform skip: rescale only when some lane's max actually grows
        const bool bump = __any(mloc > m_run);
        float alpha = 1.0f;
        if (bump) {
            const float mnew = fmaxf(m_run, mloc);
            alpha = fast_exp2(m_run - mnew);
            m_run = mnew;
        }

        float rsum = 0.0f;
#pragma unroll
        for (int t = 0; t < 4; ++t) {
            bf16x8 pk;
#pragma unroll
            for (int r = 0; r < 8; ++r) {
                const float pr = fast_exp2(st[t][r] - m_run);
                rsum += pr;
                pk[r] = (__bf16)pr;
            }
            // P^T staged row-major by q: row ln, keys 16t + 8*hf + (0..7)  (one 16B store)
            *(bf16x8*)&pbuf[ln * VPAD + 16 * t + 8 * hf] = pk;
        }
        rsum += __shfl_xor(rsum, 16, 32);
        if (bump) {
            l_run = l_run * alpha + rsum;
#pragma unroll
            for (int c = 0; c < 8; ++c) oacc[c] *= alpha;
        } else {
            l_run += rsum;
        }

        // ---- O^T += V^T * P^T   (2 key-chunks of 32 x 8 dim-chunks of 16)
#pragma unroll
        for (int g = 0; g < 2; ++g) {
            // B-frag (P^T): lane holds q=ln; element e -> key = 32g + (e<8?e:e+8) + 8*hf
            bf16x8 plo = *(const bf16x8*)&pbuf[ln * VPAD + 32 * g + 8 * hf];
            bf16x8 phi = *(const bf16x8*)&pbuf[ln * VPAD + 32 * g + 8 * hf + 16];
            v16bf pb = __builtin_shufflevector(plo, phi, 0,1,2,3,4,5,6,7,8,9,10,11,12,13,14,15);
#pragma unroll
            for (int c = 0; c < 8; ++c) {
                // A-frag (V^T): lane holds dim row 16c+ln; element e -> key = 32g + (e<8?e:e+8) + 8*hf
                const __bf16* vr = &cVT[(16 * c + ln) * VPAD + 32 * g + 8 * hf];
                bf16x8 alo = *(const bf16x8*)vr;
                bf16x8 ahi = *(const bf16x8*)(vr + 16);
                v16bf a = __builtin_shufflevector(alo, ahi, 0,1,2,3,4,5,6,7,8,9,10,11,12,13,14,15);
                oacc[c] = __builtin_amdgcn_wmma_f32_16x16x32_bf16(
                              false, a, false, pb, (short)0, oacc[c], false, false);
            }
        }
    }

    // ---- epilogue: O[q][dim] = O^T / l_run ; lane writes dims 16c+8*hf+(0..7) of row qrow
    const float inv = 1.0f / l_run;
    float* op = Og + (size_t)qrow * ROWSTRIDE + bh_off;
#pragma unroll
    for (int c = 0; c < 8; ++c) {
        f4 w0, w1;
#pragma unroll
        for (int k = 0; k < 4; ++k) { w0[k] = oacc[c][k] * inv; w1[k] = oacc[c][4 + k] * inv; }
        *(f4*)(op + 16 * c + 8 * hf)     = w0;
        *(f4*)(op + 16 * c + 8 * hf + 4) = w1;
    }
}

extern "C" void kernel_launch(void* const* d_in, const int* in_sizes, int n_in,
                              void* d_out, int out_size, void* d_ws, size_t ws_size,
                              hipStream_t stream) {
    (void)in_sizes; (void)n_in; (void)out_size; (void)d_ws; (void)ws_size;
    const float* Q = (const float*)d_in[0];
    const float* K = (const float*)d_in[1];
    const float* V = (const float*)d_in[2];
    float* O = (float*)d_out;
    dim3 grid(S_LEN / QTILE, NB, NH);
    dim3 block(256);
    hipLaunchKernelGGL(fa_fwd_causal, grid, block, 0, stream, Q, K, V, O);
}
